// ComposedBruteForceAntisymmetrizedResNet_77635828843173
// MI455X (gfx1250) — matrix operand order repaired and checked by reading, not compiled, not run on hardware
//
#include <hip/hip_runtime.h>
#include <hip/hip_bf16.h>
#include <math.h>

// ---------------------------------------------------------------------------
// Antisymmetrized ResNet, fused.  294912 rows (512 batch x 24x24 perm pairs),
// 24 -> 256 -> 256 -> 256 -> 1 tanh ResNet, signed reduction, log|sum|.
//
// Pipeline:
//   prep_weights : one-time f32 -> f16 transpose of W0/W1/W2 into d_ws,
//                  pre-padded to the LDS row stride (b128-copyable).
//   main kernel  : 128 rows/WG, 8 wave32; all GEMMs via
//                  v_wmma_f32_16x16x32_f16; weights b128-staged to LDS
//                  (W1t hoisted into the preamble so it hides under layer 0);
//                  activations ping-pong in LDS; branch-free fast tanh.
//   finalize     : sum 9 partials/batch element, log|.|.
// ---------------------------------------------------------------------------

typedef _Float16 f16;
typedef __attribute__((ext_vector_type(16))) _Float16 v16h;
typedef __attribute__((ext_vector_type(8)))  _Float16 v8h;
typedef __attribute__((ext_vector_type(8)))  float    v8f;

#define ROWS       128
#define NDENSE     256
#define HSTRIDE    264   // halves; 256 + 8 pad (row = 528B, 16B aligned, bank-skewed)
#define W0STRIDE   40    // halves; 32 + 8 pad
#define PAIRS      576
#define NBLOCKS    2304  // 294912 / 128
#define PART_PER_B 9     // 576 / 64 partial slots per batch element

// ---- LDS layout (bytes), total 305664 < 320KB -----------------------------
#define OFF_WT     0         // f16 [256][264] = 135168
#define OFF_W0T    135168    // f16 [256][40]  =  20480
#define OFF_HA     155648    // f16 [128][264] =  67584
#define OFF_HB     223232    // f16 [128][264] =  67584
#define OFF_FEATS  290816    // f16 [128][40]  =  10240
#define OFF_BIAS0  301056    // f32 [256]
#define OFF_BIAS1  302080    // f32 [256]
#define OFF_BIAS2  303104    // f32 [256]
#define OFF_WF     304128    // f32 [256]
#define OFF_RED    305152    // f32 [128]
#define SMEM_BYTES 305664

// ---- d_ws layout (bytes), total 309248 ------------------------------------
#define WS_PARTIAL 0         // f32 [4608]     =  18432
#define WS_W0T     18432     // f16 [256][40]  =  20480
#define WS_W1T     38912     // f16 [256][264] = 135168
#define WS_W2T     174080    // f16 [256][264] = 135168

// Permutations of {0,1,2,3}, lexicographic, + parity signs
__constant__ int c_P[24][4] = {
  {0,1,2,3},{0,1,3,2},{0,2,1,3},{0,2,3,1},{0,3,1,2},{0,3,2,1},
  {1,0,2,3},{1,0,3,2},{1,2,0,3},{1,2,3,0},{1,3,0,2},{1,3,2,0},
  {2,0,1,3},{2,0,3,1},{2,1,0,3},{2,1,3,0},{2,3,0,1},{2,3,1,0},
  {3,0,1,2},{3,0,2,1},{3,1,0,2},{3,1,2,0},{3,2,0,1},{3,2,1,0}
};
__constant__ float c_S[24] = {
   1.f,-1.f,-1.f, 1.f, 1.f,-1.f,
  -1.f, 1.f, 1.f,-1.f,-1.f, 1.f,
   1.f,-1.f,-1.f, 1.f, 1.f,-1.f,
  -1.f, 1.f, 1.f,-1.f,-1.f, 1.f
};

union V16 { v16h v; v8h h[2]; };

// Branch-free tanh: 1 - 2/(exp2(2*log2e*x)+1).  v_exp_f32 + v_rcp_f32,
// saturates to +/-1 (exp2 -> inf/0); no EXEC divergence, dual-issues with WMMA.
__device__ __forceinline__ float fast_tanh(float x) {
  float e = __builtin_amdgcn_exp2f(x * 2.8853900817779268f);
  return 1.f - 2.f * __builtin_amdgcn_rcpf(e + 1.f);
}

// A fragment (16x32 f16): lane L(<16) row M=L holds K = k0+[0..7], k0+16+[0..7]
__device__ __forceinline__ v16h frag_a(const f16* base, int row0, int k0,
                                       int stride, int lane) {
  int m  = row0 + (lane & 15);
  int hi = lane >> 4;
  const f16* p = base + m * stride + k0 + hi * 8;
  V16 r;
  r.h[0] = *(const v8h*)p;
  r.h[1] = *(const v8h*)(p + 16);
  return r.v;
}

// B fragment (32x16 f16) from transposed weights Wt[n][k]: lane L(<16) col
// N=L holds K = k0+[0..7], k0+16+[0..7]; lane L(>=16) col N=L-16 holds K = k0+8+[0..7], k0+24+[0..7]
__device__ __forceinline__ v16h frag_b(const f16* wt, int n0, int k0,
                                       int stride, int lane) {
  int n  = n0 + (lane & 15);
  int hi = lane >> 4;
  const f16* p = wt + n * stride + k0 + hi * 8;
  V16 r;
  r.h[0] = *(const v8h*)p;
  r.h[1] = *(const v8h*)(p + 16);
  return r.v;
}

// One 16x128 tile of D = tanh(src @ W + b) [+ src] for this wave
__device__ __forceinline__ void dense_tile(const f16* src, int sstride,
                                           const f16* wt, int wstride, int K,
                                           const float* bias, f16* dst,
                                           bool resid, int m0, int n0,
                                           int lane) {
  const v8f zero = {0.f, 0.f, 0.f, 0.f, 0.f, 0.f, 0.f, 0.f};
  v8f acc[8];
#pragma unroll
  for (int t = 0; t < 8; ++t) acc[t] = zero;

  for (int k0 = 0; k0 < K; k0 += 32) {
    v16h a = frag_a(src, m0, k0, sstride, lane);
#pragma unroll
    for (int t = 0; t < 8; ++t) {
      v16h b = frag_b(wt, n0 + t * 16, k0, wstride, lane);
      acc[t] = __builtin_amdgcn_wmma_f32_16x16x32_f16(
          false, a, false, b, (short)0, acc[t], false, false);
    }
  }

  // C/D layout: VGPR r, lanes 0-15 -> M=m0+r; lanes 16-31 -> M=m0+r+8
  const int nlo   = lane & 15;
  const int rbase = m0 + ((lane >> 4) << 3);
#pragma unroll
  for (int t = 0; t < 8; ++t) {
    int col  = n0 + t * 16 + nlo;
    float bc = bias[col];
#pragma unroll
    for (int r = 0; r < 8; ++r) {
      int row = rbase + r;
      float v = fast_tanh(acc[t][r] + bc);
      if (resid) v += (float)src[row * sstride + col];
      dst[row * HSTRIDE + col] = (f16)v;
    }
  }
}

// ---------------------------------------------------------------------------
// One-time weight prep: f32 -> f16, transposed, padded to LDS strides.
// Work items: [0,8192) W0t | [8192, +67584) W1t | [.., +67584) W2t
// ---------------------------------------------------------------------------
__global__ __launch_bounds__(256)
void prep_weights_kernel(const float* __restrict__ W0,
                         const float* __restrict__ W1,
                         const float* __restrict__ W2,
                         f16* __restrict__ w0t,
                         f16* __restrict__ w1t,
                         f16* __restrict__ w2t) {
  int q = blockIdx.x * 256 + threadIdx.x;
  if (q < 8192) {                       // W0t[n*40 + k], K padded 24 -> 32
    int k = q >> 8, n = q & 255;
    float v = (k < 24) ? W0[k * 256 + n] : 0.f;
    w0t[n * W0STRIDE + k] = (f16)v;
    return;
  }
  q -= 8192;
  const float* W = (q < 67584) ? W1 : W2;
  f16*        wt = (q < 67584) ? w1t : w2t;
  if (q >= 67584) q -= 67584;
  if (q < 65536) {                      // transpose body (coalesced reads)
    int k = q >> 8, n = q & 255;
    wt[n * HSTRIDE + k] = (f16)W[q];
  } else if (q < 67584) {               // zero the 8-half row padding
    int p = q - 65536;                  // 0..2047
    int n = p >> 3, k = 256 + (p & 7);
    wt[n * HSTRIDE + k] = (f16)0.f;
  }
}

// ---------------------------------------------------------------------------
__global__ __launch_bounds__(256)
void antisym_resnet_kernel(const float* __restrict__ x1,
                           const float* __restrict__ x2,
                           const float* __restrict__ b0,
                           const float* __restrict__ b1,
                           const float* __restrict__ b2,
                           const float* __restrict__ Wf,
                           const float* __restrict__ bf,
                           const f16* __restrict__ w0t,
                           const f16* __restrict__ w1t,
                           const f16* __restrict__ w2t,
                           float* __restrict__ partial) {
  extern __shared__ char smem[];
  f16*   Wt    = (f16*)(smem + OFF_WT);
  f16*   W0t   = (f16*)(smem + OFF_W0T);
  f16*   hA    = (f16*)(smem + OFF_HA);
  f16*   hB    = (f16*)(smem + OFF_HB);
  f16*   feats = (f16*)(smem + OFF_FEATS);
  float* bias0 = (float*)(smem + OFF_BIAS0);
  float* bias1 = (float*)(smem + OFF_BIAS1);
  float* bias2 = (float*)(smem + OFF_BIAS2);
  float* wfL   = (float*)(smem + OFF_WF);
  float* red   = (float*)(smem + OFF_RED);

  const int tid      = threadIdx.x;
  const int lane     = tid & 31;
  const int wave     = tid >> 5;
  const int m0       = wave * 16;      // 8 waves x 16 rows = 128
  const int g        = blockIdx.x;
  const int row_base = g * ROWS;

  // ---- preamble: stage W1t early (Wt is idle through layer 0), W0t,
  //      all biases, Wf, and build features ---------------------------------
  {
    const uint4* s = (const uint4*)w1t;
    uint4*       d = (uint4*)Wt;
    for (int idx = tid; idx < 135168 / 16; idx += 256) d[idx] = s[idx];
  }
  {
    const uint4* s = (const uint4*)w0t;
    uint4*       d = (uint4*)W0t;
    for (int idx = tid; idx < 20480 / 16; idx += 256) d[idx] = s[idx];
  }
  {
    int c = tid & 255;
    bias0[c] = b0[c];
    bias1[c] = b1[c];
    bias2[c] = b2[c];
    wfL[c]   = Wf[c];
  }
  for (int idx = tid; idx < ROWS * 32; idx += 256) {
    int rr  = idx >> 5;
    int c   = idx & 31;
    int gr  = row_base + rr;
    int bb  = gr / PAIRS;
    int rem = gr - bb * PAIRS;
    int i   = rem / 24;
    int j   = rem - i * 24;
    float v = 0.f;
    if (c < 12) {
      int p = c / 3, d = c - p * 3;
      v = x1[(bb * 4 + c_P[i][p]) * 3 + d];
    } else if (c < 24) {
      int cc = c - 12;
      int p = cc / 3, d = cc - p * 3;
      v = x2[(bb * 4 + c_P[j][p]) * 3 + d];
    }
    feats[rr * W0STRIDE + c] = (f16)v;
  }
  __syncthreads();

  // ---- layer 0: [128x32] @ [32x256], tanh ---------------------------------
  dense_tile(feats, W0STRIDE, W0t, W0STRIDE, 32, bias0, hA, false, m0, 0,   lane);
  dense_tile(feats, W0STRIDE, W0t, W0STRIDE, 32, bias0, hA, false, m0, 128, lane);
  __syncthreads();

  // ---- layer 1 (residual): hB = tanh(hA@W1+b1) + hA  (W1t already staged) -
  dense_tile(hA, HSTRIDE, Wt, HSTRIDE, 256, bias1, hB, true, m0, 0,   lane);
  dense_tile(hA, HSTRIDE, Wt, HSTRIDE, 256, bias1, hB, true, m0, 128, lane);
  __syncthreads();

  // ---- stage W2t (only staging left on the critical path) -----------------
  {
    const uint4* s = (const uint4*)w2t;
    uint4*       d = (uint4*)Wt;
    for (int idx = tid; idx < 135168 / 16; idx += 256) d[idx] = s[idx];
  }
  __syncthreads();

  // ---- layer 2 (residual): hA = tanh(hB@W2+b2) + hB -----------------------
  dense_tile(hB, HSTRIDE, Wt, HSTRIDE, 256, bias2, hA, true, m0, 0,   lane);
  dense_tile(hB, HSTRIDE, Wt, HSTRIDE, 256, bias2, hA, true, m0, 128, lane);
  __syncthreads();

  // ---- head: per-row dot with Wf (v8h vectorized), sign weight ------------
  if (tid < ROWS) {
    const f16* hr = hA + tid * HSTRIDE;
    float s = bf[0];
    for (int k8 = 0; k8 < 32; ++k8) {
      v8h c = *(const v8h*)(hr + k8 * 8);
      const float* w = wfL + k8 * 8;
#pragma unroll
      for (int e = 0; e < 8; ++e) s += (float)c[e] * w[e];
    }
    int gr  = row_base + tid;
    int rem = gr % PAIRS;
    int i   = rem / 24;
    int j   = rem - i * 24;
    red[tid] = s * c_S[i] * c_S[j];
  }
  __syncthreads();
  // two 64-row partial slots per block (64 | 576 keeps slots within a batch)
  if (tid < 2) {
    float acc = 0.f;
    for (int r = 0; r < 64; ++r) acc += red[tid * 64 + r];
    partial[g * 2 + tid] = acc;  // deterministic slot + order
  }
}

__global__ __launch_bounds__(256)
void antisym_finalize_kernel(const float* __restrict__ partial,
                             float* __restrict__ out) {
  int b = blockIdx.x * blockDim.x + threadIdx.x;
  if (b < 512) {
    float s = 0.f;
    for (int q = 0; q < PART_PER_B; ++q) s += partial[b * PART_PER_B + q];
    out[b] = logf(fabsf(s));
  }
}

extern "C" void kernel_launch(void* const* d_in, const int* in_sizes, int n_in,
                              void* d_out, int out_size, void* d_ws,
                              size_t ws_size, hipStream_t stream) {
  (void)in_sizes; (void)n_in; (void)out_size; (void)ws_size;
  const float* x1 = (const float*)d_in[0];
  const float* x2 = (const float*)d_in[1];
  const float* W0 = (const float*)d_in[2];
  const float* b0 = (const float*)d_in[3];
  const float* W1 = (const float*)d_in[4];
  const float* b1 = (const float*)d_in[5];
  const float* W2 = (const float*)d_in[6];
  const float* b2 = (const float*)d_in[7];
  const float* Wf = (const float*)d_in[8];
  const float* bf = (const float*)d_in[9];

  char* ws        = (char*)d_ws;
  float* partial  = (float*)(ws + WS_PARTIAL);
  f16*   w0t      = (f16*)(ws + WS_W0T);
  f16*   w1t      = (f16*)(ws + WS_W1T);
  f16*   w2t      = (f16*)(ws + WS_W2T);
  float* out      = (float*)d_out;

  (void)hipFuncSetAttribute((const void*)antisym_resnet_kernel,
                            hipFuncAttributeMaxDynamicSharedMemorySize,
                            (int)SMEM_BYTES);

  // (8192 + 67584 + 67584) work items = 560 blocks of 256
  prep_weights_kernel<<<560, 256, 0, stream>>>(W0, W1, W2, w0t, w1t, w2t);
  antisym_resnet_kernel<<<NBLOCKS, 256, SMEM_BYTES, stream>>>(
      x1, x2, b0, b1, b2, Wf, bf, w0t, w1t, w2t, partial);
  antisym_finalize_kernel<<<2, 256, 0, stream>>>(partial, out);
}
